// GINRegression_72215580115596
// MI455X (gfx1250) — matrix operand order repaired and checked by reading, not compiled, not run on hardware
//
#include <hip/hip_runtime.h>
#include <hip/hip_bf16.h>

typedef __attribute__((ext_vector_type(16))) _Float16 v16h;
typedef __attribute__((ext_vector_type(8)))  _Float16 v8h;
typedef __attribute__((ext_vector_type(8)))  float    v8f;

#define D_FEAT 64
#define WAVES_PER_BLOCK 10   // 320 threads; 100000/16 = 6250 tiles = 625 blocks * 10 waves exactly

// ---------------------------------------------------------------------------
// Scatter-add aggregation: agg[dst] += h[src], one thread per (edge, feature).
// Node table (25.6MB) is L2-resident on MI455X (192MB L2), so the random
// gather + atomics stay on-chip.
// ---------------------------------------------------------------------------
__global__ void gin_agg_kernel(const float* __restrict__ h,
                               float* __restrict__ agg,
                               const int* __restrict__ src,
                               const int* __restrict__ dst,
                               long long total) {
    long long id = (long long)blockIdx.x * blockDim.x + threadIdx.x;
    if (id >= total) return;
    int e = (int)(id >> 6);
    int d = (int)(id & 63);
    int s = src[e];
    int t = dst[e];
    atomicAdd(&agg[(size_t)t * D_FEAT + d], h[(size_t)s * D_FEAT + d]);
}

// ---------------------------------------------------------------------------
// Fused GIN MLP:  hout = relu(gamma * (relu((h+agg) @ W1 + b1) @ W2 + b2) + beta)
// One wave per 16-node tile; f16 WMMA with f32 accumulate.
// ---------------------------------------------------------------------------
__global__ __launch_bounds__(32 * WAVES_PER_BLOCK)
void gin_mlp_kernel(const float* __restrict__ hin,
                    const float* agg,            // may alias hout (in-place)
                    float* hout,
                    const float* __restrict__ W1, const float* __restrict__ b1,
                    const float* __restrict__ W2, const float* __restrict__ b2,
                    const float* __restrict__ gamma, const float* __restrict__ beta,
                    int nTiles) {
    // Weights stored transposed in LDS as f16: wT[n*64 + k] = W[k*64 + n]
    __shared__ alignas(32) _Float16 w1t[64 * 64];
    __shared__ alignas(32) _Float16 w2t[64 * 64];
    __shared__ alignas(32) _Float16 mid[WAVES_PER_BLOCK][16 * 64];

    for (int i = threadIdx.x; i < 64 * 64; i += blockDim.x) {
        int k = i >> 6, n = i & 63;
        w1t[n * 64 + k] = (_Float16)W1[k * 64 + n];
        w2t[n * 64 + k] = (_Float16)W2[k * 64 + n];
    }
    __syncthreads();

    const int wave   = threadIdx.x >> 5;
    const int lane   = threadIdx.x & 31;
    const int laneM  = lane & 15;     // row (A) or column (B/C) within 16
    const int hiHalf = lane >> 4;     // which half-wave group
    const int tile   = blockIdx.x * WAVES_PER_BLOCK + wave;
    const bool active = (tile < nTiles);   // wave-uniform

    _Float16* m = mid[wave];
    v8f acc[4] = {};

    if (active) {
        const int node = tile * 16 + laneM;
        const float4* h4 = (const float4*)(hin + (size_t)node * D_FEAT);
        const float4* g4 = (const float4*)(agg + (size_t)node * D_FEAT);

        // ---- GEMM1: (h + agg) @ W1, K = 64 in two k-steps of 32 ----
        #pragma unroll
        for (int ks = 0; ks < 2; ++ks) {
            const int kb = ks * 32 + hiHalf * 16;
            v16h afrag;
            #pragma unroll
            for (int c = 0; c < 2; ++c) {
                const int f4 = (kb + c * 8) >> 2;
                float4 x0 = h4[f4], x1 = h4[f4 + 1];
                float4 y0 = g4[f4], y1 = g4[f4 + 1];
                const int o = c * 8;
                afrag[o + 0] = (_Float16)(x0.x + y0.x);
                afrag[o + 1] = (_Float16)(x0.y + y0.y);
                afrag[o + 2] = (_Float16)(x0.z + y0.z);
                afrag[o + 3] = (_Float16)(x0.w + y0.w);
                afrag[o + 4] = (_Float16)(x1.x + y1.x);
                afrag[o + 5] = (_Float16)(x1.y + y1.y);
                afrag[o + 6] = (_Float16)(x1.z + y1.z);
                afrag[o + 7] = (_Float16)(x1.w + y1.w);
            }
            const int krow = ks * 32 + hiHalf * 16;
            #pragma unroll
            for (int nt = 0; nt < 4; ++nt) {
                const int col = nt * 16 + laneM;
                v16h bfrag = *(const v16h*)(&w1t[col * 64 + krow]);
                acc[nt] = __builtin_amdgcn_wmma_f32_16x16x32_f16(
                    false, afrag, false, bfrag, (short)0, acc[nt], false, false);
            }
        }

        // ---- bias + ReLU, spill 16x64 intermediate to LDS as f16 ----
        #pragma unroll
        for (int nt = 0; nt < 4; ++nt) {
            const int col = nt * 16 + laneM;
            const float bb = b1[col];
            #pragma unroll
            for (int r = 0; r < 8; ++r) {
                float v = acc[nt][r] + bb;
                v = v > 0.0f ? v : 0.0f;
                m[(r + hiHalf * 8) * 64 + col] = (_Float16)v;
            }
        }
    }
    __syncthreads();

    if (active) {
        // ---- GEMM2: mid @ W2 ----
        v8f acc2[4] = {};
        #pragma unroll
        for (int ks = 0; ks < 2; ++ks) {
            const int kb = ks * 32 + hiHalf * 16;
            v16h afrag;
            ((v8h*)&afrag)[0] = *(const v8h*)(&m[laneM * 64 + kb]);
            ((v8h*)&afrag)[1] = *(const v8h*)(&m[laneM * 64 + kb + 8]);
            const int krow = ks * 32 + hiHalf * 16;
            #pragma unroll
            for (int nt = 0; nt < 4; ++nt) {
                const int col = nt * 16 + laneM;
                v16h bfrag = *(const v16h*)(&w2t[col * 64 + krow]);
                acc2[nt] = __builtin_amdgcn_wmma_f32_16x16x32_f16(
                    false, afrag, false, bfrag, (short)0, acc2[nt], false, false);
            }
        }

        // ---- bias + BN affine + ReLU, store f32 ----
        #pragma unroll
        for (int nt = 0; nt < 4; ++nt) {
            const int col = nt * 16 + laneM;
            const float bb = b2[col];
            const float gm = gamma[col];
            const float bt = beta[col];
            #pragma unroll
            for (int r = 0; r < 8; ++r) {
                float v = acc2[nt][r] + bb;
                v = v * gm + bt;
                v = v > 0.0f ? v : 0.0f;
                hout[(size_t)(tile * 16 + r + hiHalf * 8) * D_FEAT + col] = v;
            }
        }
    }
}

// ---------------------------------------------------------------------------
// Segment-sum pooling: pooled[batch[n]] += h[n]; counts[batch[n]] += 1
// ---------------------------------------------------------------------------
__global__ void gin_pool_kernel(const float* __restrict__ h,
                                const int* __restrict__ batch,
                                float* __restrict__ pooled,
                                float* __restrict__ counts,
                                long long total) {
    long long id = (long long)blockIdx.x * blockDim.x + threadIdx.x;
    if (id >= total) return;
    int n = (int)(id >> 6);
    int d = (int)(id & 63);
    int g = batch[n];
    atomicAdd(&pooled[(size_t)g * D_FEAT + d], h[(size_t)n * D_FEAT + d]);
    if (d == 0) atomicAdd(&counts[g], 1.0f);
}

// ---------------------------------------------------------------------------
// Head: out[g] = relu(mean_g @ fc1 + b1) @ fc2 + b2   (one thread per graph)
// ---------------------------------------------------------------------------
__global__ void gin_head_kernel(const float* __restrict__ pooled,
                                const float* __restrict__ counts,
                                const float* __restrict__ fc1W,
                                const float* __restrict__ fc1b,
                                const float* __restrict__ fc2W,
                                const float* __restrict__ fc2b,
                                float* __restrict__ out, int G) {
    int g = blockIdx.x * blockDim.x + threadIdx.x;
    if (g >= G) return;
    float c = counts[g];
    c = c > 1.0f ? c : 1.0f;
    const float inv = 1.0f / c;
    float p[64];
    #pragma unroll
    for (int k = 0; k < 64; ++k) p[k] = pooled[(size_t)g * 64 + k] * inv;
    float acc = fc2b[0];
    for (int j = 0; j < 32; ++j) {
        float s = fc1b[j];
        #pragma unroll
        for (int k = 0; k < 64; ++k) s += p[k] * fc1W[k * 32 + j];
        s = s > 0.0f ? s : 0.0f;
        acc += s * fc2W[j];
    }
    out[g] = acc;
}

extern "C" void kernel_launch(void* const* d_in, const int* in_sizes, int n_in,
                              void* d_out, int out_size, void* d_ws, size_t ws_size,
                              hipStream_t stream) {
    (void)n_in; (void)ws_size;
    const float* x     = (const float*)d_in[0];
    const int*   ei    = (const int*)d_in[1];
    const int*   batch = (const int*)d_in[2];
    const float* W1    = (const float*)d_in[3];
    const float* b1    = (const float*)d_in[4];
    const float* W2    = (const float*)d_in[5];
    const float* b2    = (const float*)d_in[6];
    const float* gamma = (const float*)d_in[7];
    const float* beta  = (const float*)d_in[8];
    const float* fc1W  = (const float*)d_in[9];
    const float* fc1b  = (const float*)d_in[10];
    const float* fc2W  = (const float*)d_in[11];
    const float* fc2b  = (const float*)d_in[12];
    float* out = (float*)d_out;

    const int N = in_sizes[0] / D_FEAT;
    const int E = in_sizes[1] / 2;
    const int G = out_size;
    const int* src = ei;
    const int* dst = ei + E;

    float* bufA   = (float*)d_ws;                 // N*64
    float* bufB   = bufA + (size_t)N * D_FEAT;    // N*64
    float* pooled = bufB + (size_t)N * D_FEAT;    // G*64
    float* counts = pooled + (size_t)G * D_FEAT;  // G

    const long long aggTot = (long long)E * D_FEAT;
    const unsigned aggBlocks = (unsigned)((aggTot + 255) / 256);
    const int nTiles = (N + 15) / 16;
    const unsigned mlpBlocks = (unsigned)((nTiles + WAVES_PER_BLOCK - 1) / WAVES_PER_BLOCK);
    const size_t featBytes = (size_t)N * D_FEAT * sizeof(float);

    // ---- Layer 0: x -> bufB (agg in bufA) ----
    hipMemsetAsync(bufA, 0, featBytes, stream);
    gin_agg_kernel<<<aggBlocks, 256, 0, stream>>>(x, bufA, src, dst, aggTot);
    gin_mlp_kernel<<<mlpBlocks, 32 * WAVES_PER_BLOCK, 0, stream>>>(
        x, bufA, bufB, W1, b1, W2, b2, gamma, beta, nTiles);

    // ---- Layer 1: bufB -> bufA (agg in bufA, in-place) ----
    hipMemsetAsync(bufA, 0, featBytes, stream);
    gin_agg_kernel<<<aggBlocks, 256, 0, stream>>>(bufB, bufA, src, dst, aggTot);
    gin_mlp_kernel<<<mlpBlocks, 32 * WAVES_PER_BLOCK, 0, stream>>>(
        bufB, bufA, bufA, W1 + 4096, b1 + 64, W2 + 4096, b2 + 64,
        gamma + 64, beta + 64, nTiles);

    // ---- Layer 2: bufA -> bufB (agg in bufB, in-place) ----
    hipMemsetAsync(bufB, 0, featBytes, stream);
    gin_agg_kernel<<<aggBlocks, 256, 0, stream>>>(bufA, bufB, src, dst, aggTot);
    gin_mlp_kernel<<<mlpBlocks, 32 * WAVES_PER_BLOCK, 0, stream>>>(
        bufA, bufB, bufB, W1 + 8192, b1 + 128, W2 + 8192, b2 + 128,
        gamma + 128, beta + 128, nTiles);

    // ---- Mean pool + head ----
    hipMemsetAsync(pooled, 0, ((size_t)G * D_FEAT + G) * sizeof(float), stream);
    const long long poolTot = (long long)N * D_FEAT;
    gin_pool_kernel<<<(unsigned)((poolTot + 255) / 256), 256, 0, stream>>>(
        bufB, batch, pooled, counts, poolTot);
    gin_head_kernel<<<1, 128, 0, stream>>>(pooled, counts, fc1W, fc1b, fc2W, fc2b, out, G);
}
